// GAT_32719060861315
// MI455X (gfx1250) — matrix operand order repaired and checked by reading, not compiled, not run on hardware
//
#include <hip/hip_runtime.h>
#include <math.h>

// ---------------------------------------------------------------------------
// GAT forward for MI455X (gfx1250, wave32, WMMA).
// Dense node GEMMs: v_wmma_f32_16x16x32_bf16, weights staged in LDS via
// global_load_async_to_lds_b128 (ASYNCcnt), fragments via ds_load_b128.
// Edge softmax/scatter phase uses f32 global atomics (k/q/v fit in 192MB L2).
// ---------------------------------------------------------------------------

typedef __bf16 bf16;
typedef __attribute__((ext_vector_type(16))) __bf16 v16bf;
typedef __attribute__((ext_vector_type(8)))  __bf16 v8bf;
typedef __attribute__((ext_vector_type(8)))  float  v8f;
typedef __attribute__((ext_vector_type(4)))  float  v4f;

static constexpr int NN = 100000;   // nodes
static constexpr int EE = 1600000;  // edges
static constexpr int DD = 128;      // layer width
static constexpr int HH = 2;        // heads
static constexpr int DK = 64;       // per-head dim
static constexpr int LDS_STRIDE = 136;  // padded row (bf16 elems): 272B -> no bank conflicts

__device__ __forceinline__ float gelu_tanh(float x) {
    float x3 = x * x * x;
    float t = tanhf(0.7978845608028654f * (x + 0.044715f * x3));
    return 0.5f * x * (1.0f + t);
}

__device__ __forceinline__ void atomicMaxFloat(float* addr, float v) {
    // smax initialized to -inf; int ordering matches float ordering for
    // non-negative values, uint-min matches for negative values.
    if (v >= 0.f) atomicMax((int*)addr, __float_as_int(v));
    else          atomicMin((unsigned int*)addr, __float_as_uint(v));
}

// ---------------------------------------------------------------------------
// Weight pack: f32 row-major W[k][n] (128x128) -> bf16 transposed Bt[n][k]
// so that WMMA B-fragments are two contiguous 8-element k runs per lane.
// ---------------------------------------------------------------------------
__global__ void pack_w_bf16_T(const float* __restrict__ W, bf16* __restrict__ Bt) {
    int id = blockIdx.x * 256 + threadIdx.x;   // 0..16383
    if (id >= DD * DD) return;
    int n = id >> 7, k = id & 127;
    Bt[id] = (bf16)W[k * DD + n];
}

// ---------------------------------------------------------------------------
// WMMA GEMM: out[nrows x 128] = A(bf16)[nrows x 128] @ Bt^T + bias (+addend)
// (+gelu). Optionally also writes a bf16 copy of the output.
// Block = 256 threads = 8 waves; wave w handles rows [blk*128 + w*16, +16).
// Bt (32KB) staged once per block into LDS via async-to-LDS, then each wave
// reads fragments with ds_load_b128. 32 wmma per wave.
// nrows must be a multiple of 16 (100000 is).
// ---------------------------------------------------------------------------
__global__ __launch_bounds__(256)
void gemm128_wmma(const bf16* __restrict__ A, const bf16* __restrict__ Bt,
                  const float* __restrict__ bias, const float* __restrict__ addend,
                  float* __restrict__ outF, bf16* __restrict__ outB,
                  int nrows, int gelu_flag)
{
    __shared__ __align__(16) bf16 ldsB[DD * LDS_STRIDE];   // 34,816 bytes

    const int tid  = threadIdx.x;
    const int lane = tid & 31;
    const int wave = tid >> 5;

    // ---- stage Bt into LDS (padded rows) with async copies: 2048 x 16B ----
    #pragma unroll
    for (int i = 0; i < 8; ++i) {
        const int chunk = tid + i * 256;           // 0..2047
        const int n   = chunk >> 4;                // source row (column of W)
        const int sub = chunk & 15;                // 16B chunk within row
        unsigned lds_addr =
            (unsigned)(uintptr_t)(&ldsB[n * LDS_STRIDE + sub * 8]);
        unsigned long long gaddr =
            (unsigned long long)(uintptr_t)(Bt + (size_t)chunk * 8);
        asm volatile("global_load_async_to_lds_b128 %0, %1, off"
                     :: "v"(lds_addr), "v"(gaddr) : "memory");
    }
    asm volatile("s_wait_asynccnt 0x0" ::: "memory");
    __syncthreads();

    const int rowBase = blockIdx.x * 128 + wave * 16;
    if (rowBase + 16 > nrows) return;   // whole-wave guard (after the barrier)

    const int m    = lane & 15;         // A row within tile / B column
    const int half = lane >> 4;         // lane-group (K split)

    v8f acc[8] = {};                    // 8 tiles x 16x16 f32 accum

    const bf16* arow = A + (size_t)(rowBase + m) * DD;

    #pragma unroll
    for (int k0 = 0; k0 < DD; k0 += 32) {
        // A fragment 16x32: lane holds K in [k0+half*8, +8) and [k0+16+half*8, +8)
        union { v16bf v; v8bf h[2]; } a;
        a.h[0] = *(const v8bf*)(arow + k0 + half * 8);
        a.h[1] = *(const v8bf*)(arow + k0 + 16 + half * 8);
        #pragma unroll
        for (int j = 0; j < 8; ++j) {
            // B fragment 32x16 for columns [j*16, j*16+16): lane (col=j*16+m)
            // holds K in [k0+half*8, +8) and [k0+16+half*8, +8) of the padded LDS row.
            const bf16* bp = &ldsB[(j * 16 + m) * LDS_STRIDE + k0 + half * 8];
            union { v16bf v; v8bf h[2]; } b;
            b.h[0] = *(const v8bf*)(bp);
            b.h[1] = *(const v8bf*)(bp + 16);
            acc[j] = __builtin_amdgcn_wmma_f32_16x16x32_bf16(
                false, a.v, false, b.v, (short)0, acc[j], false, false);
        }
    }

    // Epilogue: C/D layout -> VGPR r, lane: row = half*8 + r, col = m
    #pragma unroll
    for (int j = 0; j < 8; ++j) {
        const int col = j * 16 + m;
        const float bv = bias ? bias[col] : 0.f;
        #pragma unroll
        for (int r = 0; r < 8; ++r) {
            const size_t idx = (size_t)(rowBase + half * 8 + r) * DD + col;
            float val = acc[j][r] + bv;
            if (addend)    val += addend[idx];
            if (gelu_flag) val = gelu_tanh(val);
            outF[idx] = val;
            if (outB) outB[idx] = (bf16)val;
        }
    }
}

// ---------------------------------------------------------------------------
// fc0: h = x[N,16] @ W[16,128] + b   (K=16 too skinny for WMMA)
// ---------------------------------------------------------------------------
__global__ void fc0_kernel(const float* __restrict__ x, const float* __restrict__ w,
                           const float* __restrict__ b,
                           float* __restrict__ hF, bf16* __restrict__ hB)
{
    size_t id = (size_t)blockIdx.x * 256 + threadIdx.x;
    if (id >= (size_t)NN * DD) return;
    int d = (int)(id & 127);
    size_t node = id >> 7;
    const float* xr = x + node * 16;
    float s = b[d];
    #pragma unroll
    for (int i = 0; i < 16; ++i) s += xr[i] * w[i * DD + d];
    hF[id] = s;
    hB[id] = (bf16)s;
}

__global__ void fill_f32(float* __restrict__ p, float v, size_t n) {
    size_t i = (size_t)blockIdx.x * 256 + threadIdx.x;
    if (i < n) p[i] = v;
}

// ---------------------------------------------------------------------------
// Edge phase. One thread per (edge, head); k/q/v live in L2 (154MB < 192MB).
// ---------------------------------------------------------------------------
__global__ void edge_score(const int* __restrict__ src, const int* __restrict__ dst,
                           const float* __restrict__ kF, const float* __restrict__ qF,
                           float* __restrict__ score, float* __restrict__ smax)
{
    int t = blockIdx.x * 256 + threadIdx.x;
    if (t >= EE * HH) return;
    int e = t >> 1, h = t & 1;
    int s = src[e], d = dst[e];
    const v4f* kp = (const v4f*)(kF + (size_t)s * DD + h * DK);
    const v4f* qp = (const v4f*)(qF + (size_t)d * DD + h * DK);
    float acc = 0.f;
    #pragma unroll
    for (int i = 0; i < DK / 4; ++i) {
        v4f a = kp[i], q = qp[i];
        acc += a.x * q.x + a.y * q.y + a.z * q.z + a.w * q.w;
    }
    score[t] = acc;
    atomicMaxFloat(&smax[d * HH + h], acc);
}

__global__ void edge_exp(const int* __restrict__ dst,
                         float* __restrict__ score,          // in: score, out: exp
                         const float* __restrict__ smax,
                         float* __restrict__ denom)
{
    int t = blockIdx.x * 256 + threadIdx.x;
    if (t >= EE * HH) return;
    int e = t >> 1, h = t & 1;
    int d = dst[e];
    float ex = __expf(score[t] - smax[d * HH + h]);
    score[t] = ex;
    atomicAdd(&denom[d * HH + h], ex);
}

__global__ void edge_scatter(const int* __restrict__ src, const int* __restrict__ dst,
                             const float* __restrict__ ex, const float* __restrict__ denom,
                             const float* __restrict__ vF, float* __restrict__ res)
{
    int t = blockIdx.x * 256 + threadIdx.x;
    if (t >= EE * HH) return;
    int e = t >> 1, h = t & 1;
    int s = src[e], d = dst[e];
    float w = ex[t] / denom[d * HH + h];
    const v4f* vp = (const v4f*)(vF + (size_t)s * DD + h * DK);
    float* rp = res + (size_t)d * DD + h * DK;
    #pragma unroll
    for (int i = 0; i < DK / 4; ++i) {
        v4f val = vp[i];
        atomicAdd(rp + i * 4 + 0, val.x * w);
        atomicAdd(rp + i * 4 + 1, val.y * w);
        atomicAdd(rp + i * 4 + 2, val.z * w);
        atomicAdd(rp + i * 4 + 3, val.w * w);
    }
}

// ---------------------------------------------------------------------------
// fc2: out = h[N,128] @ W[128,4] + b   (N=4 too skinny for WMMA)
// ---------------------------------------------------------------------------
__global__ void fc2_kernel(const float* __restrict__ h, const float* __restrict__ w,
                           const float* __restrict__ b, float* __restrict__ out)
{
    int n = blockIdx.x * 256 + threadIdx.x;
    if (n >= NN) return;
    float s0 = b[0], s1 = b[1], s2 = b[2], s3 = b[3];
    const float* hr = h + (size_t)n * DD;
    #pragma unroll 4
    for (int k = 0; k < DD; ++k) {
        float hv = hr[k];
        const float* wr = w + k * 4;
        s0 += hv * wr[0]; s1 += hv * wr[1]; s2 += hv * wr[2]; s3 += hv * wr[3];
    }
    out[(size_t)n * 4 + 0] = s0; out[(size_t)n * 4 + 1] = s1;
    out[(size_t)n * 4 + 2] = s2; out[(size_t)n * 4 + 3] = s3;
}

// ---------------------------------------------------------------------------
extern "C" void kernel_launch(void* const* d_in, const int* in_sizes, int n_in,
                              void* d_out, int out_size, void* d_ws, size_t ws_size,
                              hipStream_t stream)
{
    const float* x     = (const float*)d_in[0];
    const int*   src   = (const int*)  d_in[1];
    const int*   dst   = (const int*)  d_in[2];
    const float* fc0_w = (const float*)d_in[3];
    const float* fc0_b = (const float*)d_in[4];
    const float* Wk    = (const float*)d_in[5];
    const float* bk    = (const float*)d_in[6];
    const float* Wq    = (const float*)d_in[7];
    const float* bq    = (const float*)d_in[8];
    const float* Wv    = (const float*)d_in[9];
    const float* bv    = (const float*)d_in[10];
    const float* ws_w  = (const float*)d_in[11];
    const float* ws_b  = (const float*)d_in[12];
    const float* fc1_w = (const float*)d_in[13];
    const float* fc1_b = (const float*)d_in[14];
    const float* fc2_w = (const float*)d_in[15];
    const float* fc2_b = (const float*)d_in[16];
    float* out = (float*)d_out;

    // ---- workspace carve-up (256B aligned) ----
    char* ws = (char*)d_ws;
    size_t off = 0;
    auto carve = [&](size_t bytes) -> void* {
        void* p = ws + off;
        off += (bytes + 255) & ~(size_t)255;
        return p;
    };
    float* hF   = (float*)carve((size_t)NN * DD * 4);
    bf16*  hB   = (bf16*) carve((size_t)NN * DD * 2);
    float* kF   = (float*)carve((size_t)NN * DD * 4);
    float* qF   = (float*)carve((size_t)NN * DD * 4);
    float* vF   = (float*)carve((size_t)NN * DD * 4);
    float* resF = (float*)carve((size_t)NN * DD * 4);
    float* scoreB = (float*)carve((size_t)EE * HH * 4);   // reused as exp buffer
    float* smax   = (float*)carve((size_t)NN * HH * 4);
    float* denom  = (float*)carve((size_t)NN * HH * 4);
    bf16* pk[2], *pq[2], *pv[2], *ps[2], *pfc1;
    for (int l = 0; l < 2; ++l) {
        pk[l] = (bf16*)carve((size_t)DD * DD * 2);
        pq[l] = (bf16*)carve((size_t)DD * DD * 2);
        pv[l] = (bf16*)carve((size_t)DD * DD * 2);
        ps[l] = (bf16*)carve((size_t)DD * DD * 2);
    }
    pfc1 = (bf16*)carve((size_t)DD * DD * 2);
    (void)ws_size; (void)n_in; (void)in_sizes; (void)out_size;

    const dim3 blk(256);
    const int gridPack = (DD * DD + 255) / 256;
    const int gridGemm = (NN + 127) / 128;
    const int gridElem = (int)(((size_t)NN * DD + 255) / 256);
    const int gridEdge = (EE * HH + 255) / 256;
    const int gridNH   = (NN * HH + 255) / 256;
    const int gridNode = (NN + 255) / 256;

    // ---- pack all weights to bf16 transposed layout ----
    for (int l = 0; l < 2; ++l) {
        pack_w_bf16_T<<<gridPack, blk, 0, stream>>>(Wk   + (size_t)l * DD * DD, pk[l]);
        pack_w_bf16_T<<<gridPack, blk, 0, stream>>>(Wq   + (size_t)l * DD * DD, pq[l]);
        pack_w_bf16_T<<<gridPack, blk, 0, stream>>>(Wv   + (size_t)l * DD * DD, pv[l]);
        pack_w_bf16_T<<<gridPack, blk, 0, stream>>>(ws_w + (size_t)l * DD * DD, ps[l]);
    }
    pack_w_bf16_T<<<gridPack, blk, 0, stream>>>(fc1_w, pfc1);

    // ---- fc0 ----
    fc0_kernel<<<gridElem, blk, 0, stream>>>(x, fc0_w, fc0_b, hF, hB);

    // ---- GAT layers ----
    for (int l = 0; l < 2; ++l) {
        gemm128_wmma<<<gridGemm, blk, 0, stream>>>(hB, pk[l], bk + l * DD, nullptr,
                                                   kF, nullptr, NN, 0);
        gemm128_wmma<<<gridGemm, blk, 0, stream>>>(hB, pq[l], bq + l * DD, nullptr,
                                                   qF, nullptr, NN, 0);
        gemm128_wmma<<<gridGemm, blk, 0, stream>>>(hB, pv[l], bv + l * DD, nullptr,
                                                   vF, nullptr, NN, 0);

        fill_f32<<<gridNH,   blk, 0, stream>>>(smax, -INFINITY, (size_t)NN * HH);
        fill_f32<<<gridNH,   blk, 0, stream>>>(denom, 0.f,      (size_t)NN * HH);
        fill_f32<<<gridElem, blk, 0, stream>>>(resF,  0.f,      (size_t)NN * DD);

        edge_score  <<<gridEdge, blk, 0, stream>>>(src, dst, kF, qF, scoreB, smax);
        edge_exp    <<<gridEdge, blk, 0, stream>>>(dst, scoreB, smax, denom);
        edge_scatter<<<gridEdge, blk, 0, stream>>>(src, dst, scoreB, denom, vF, resF);

        // h = (h @ ws + ws_b) + res  (+gelu except last layer); updates hF/hB in place
        gemm128_wmma<<<gridGemm, blk, 0, stream>>>(hB, ps[l], ws_b + l * DD, resF,
                                                   hF, hB, NN, (l == 0) ? 1 : 0);
    }

    // ---- fc1 (gelu) into hF, then fc2 ----
    gemm128_wmma<<<gridGemm, blk, 0, stream>>>(hB, pfc1, fc1_b, nullptr,
                                               hF, nullptr, NN, 1);
    fc2_kernel<<<gridNode, blk, 0, stream>>>(hF, fc2_w, fc2_b, out);
}
